// GraphConvolution_70566312673379
// MI455X (gfx1250) — matrix operand order repaired and checked by reading, not compiled, not run on hardware
//
#include <hip/hip_runtime.h>

typedef __attribute__((ext_vector_type(16))) __bf16 v16bf;
typedef __attribute__((ext_vector_type(8)))  float  v8f;

#define N_NODES 100000
#define DIN     512
#define DOUT    256
#define NEDGE   3200000
#define KEEP    0.9f

// ---------------------------------------------------------------------------
// JAX threefry2x32 (20 rounds), key = jax.random.key(42) -> (0, 42)
// ---------------------------------------------------------------------------
__device__ __forceinline__ unsigned rotl32(unsigned x, int n) {
    return (x << n) | (x >> (32 - n));
}

__device__ __forceinline__ void threefry2x32(unsigned ks0, unsigned ks1,
                                             unsigned x0, unsigned x1,
                                             unsigned& o0, unsigned& o1) {
    unsigned ks2 = ks0 ^ ks1 ^ 0x1BD11BDAu;
#define TF_RND(r) { x0 += x1; x1 = rotl32(x1, r); x1 ^= x0; }
    x0 += ks0; x1 += ks1;
    TF_RND(13) TF_RND(15) TF_RND(26) TF_RND(6)
    x0 += ks1; x1 += ks2 + 1u;
    TF_RND(17) TF_RND(29) TF_RND(16) TF_RND(24)
    x0 += ks2; x1 += ks0 + 2u;
    TF_RND(13) TF_RND(15) TF_RND(26) TF_RND(6)
    x0 += ks0; x1 += ks1 + 3u;
    TF_RND(17) TF_RND(29) TF_RND(16) TF_RND(24)
    x0 += ks1; x1 += ks2 + 4u;
    TF_RND(13) TF_RND(15) TF_RND(26) TF_RND(6)
    x0 += ks2; x1 += ks0 + 5u;
#undef TF_RND
    o0 = x0; o1 = x1;
}

// dropout(x)[g] with JAX bernoulli(key(42), 0.9) mask over flat index g
__device__ __forceinline__ float dropout_val(const float* __restrict__ x, unsigned g) {
    const unsigned H = (N_NODES * DIN) / 2u;   // 25,600,000 (even split)
    unsigned c0, c1, o0, o1;
    bool lo = (g < H);
    if (lo) { c0 = g;     c1 = g + H; }
    else    { c0 = g - H; c1 = g;     }
    threefry2x32(0u, 42u, c0, c1, o0, o1);
    unsigned bits = lo ? o0 : o1;
    float u = (float)(bits >> 9) * (1.0f / 8388608.0f);   // [0,1) with 23 mantissa bits
    float v = x[g];
    return (u < KEEP) ? v * (1.0f / KEEP) : 0.0f;
}

// ---------------------------------------------------------------------------
// Pack W [512,256] f32 row-major -> bf16 WMMA B-fragments.
// Fragment (kc, nt): lane = (kr>=16)*16 + (n%16) holds 16 contiguous bf16,
// j = kr%16  (K = kc*32 + (lane/16)*16 + j), same contiguous K order as the A fragment.
// ---------------------------------------------------------------------------
__global__ void pack_w_kernel(const float* __restrict__ W, __bf16* __restrict__ Wp) {
    int i = blockIdx.x * 256 + threadIdx.x;
    if (i >= DIN * DOUT) return;
    int k = i / DOUT, n = i % DOUT;
    int kc = k >> 5, kr = k & 31;
    int nt = n >> 4, nl = n & 15;
    int lane = ((kr >> 4) << 4) + nl;
    int j = kr & 15;
    Wp[(((kc * 16 + nt) * 32) + lane) * 16 + j] = (__bf16)W[i];
}

// ---------------------------------------------------------------------------
// Fused dropout + GEMM: h[16 rows x 256 cols] per block via v_wmma_f32_16x16x32_bf16
// ---------------------------------------------------------------------------
__global__ void __launch_bounds__(256)
gemm_dropout_kernel(const float* __restrict__ x,
                    const __bf16* __restrict__ Wp,
                    float* __restrict__ h) {
    __shared__ __bf16 sA[16 * DIN];          // 16 KB of the 320 KB WGP LDS
    const int m0 = blockIdx.x * 16;

    // Stage 16x512 A panel with dropout applied, f32 -> bf16 (each x elem once)
    for (int t = threadIdx.x; t < 16 * DIN; t += 256) {
        int row = t >> 9, k = t & (DIN - 1);
        unsigned g = (unsigned)(m0 + row) * DIN + k;
        sA[t] = (__bf16)dropout_val(x, g);
    }
    __syncthreads();

    const int wave = threadIdx.x >> 5;
    const int lane = threadIdx.x & 31;
    const int half = lane >> 4;              // K-half select
    const int mrow = lane & 15;              // A row for this lane
    const int nt0 = wave * 2, nt1 = wave * 2 + 1;

    v8f acc0 = {}; v8f acc1 = {};

    for (int kc = 0; kc < DIN / 32; ++kc) {
        // A fragment (contiguous K order, K = half*16 + j): one contiguous 32B run from LDS
        v16bf a;
        const __bf16* ap = &sA[mrow * DIN + kc * 32 + half * 16];
#pragma unroll
        for (int j = 0; j < 8; ++j) a[j] = ap[j];
#pragma unroll
        for (int j = 0; j < 8; ++j) a[8 + j] = ap[8 + j];

        // B fragments: pre-swizzled, 32B contiguous per lane
        v16bf b0 = *(const v16bf*)&Wp[(((kc * 16 + nt0) * 32) + lane) * 16];
        v16bf b1 = *(const v16bf*)&Wp[(((kc * 16 + nt1) * 32) + lane) * 16];

        acc0 = __builtin_amdgcn_wmma_f32_16x16x32_bf16(false, a, false, b0,
                                                       (short)0, acc0, false, false);
        acc1 = __builtin_amdgcn_wmma_f32_16x16x32_bf16(false, a, false, b1,
                                                       (short)0, acc1, false, false);
    }

    // C/D layout: VGPR r, lane L -> M = r + 8*(L>=16), N = L%16
    const int col = lane & 15;
    const int rofs = half * 8;
#pragma unroll
    for (int r = 0; r < 8; ++r) {
        h[(size_t)(m0 + rofs + r) * DOUT + nt0 * 16 + col] = acc0[r];
        h[(size_t)(m0 + rofs + r) * DOUT + nt1 * 16 + col] = acc1[r];
    }
}

// ---------------------------------------------------------------------------
// Edge scatter: msgs = w_e * h[src]; atomic accumulate into out[dst]
// ---------------------------------------------------------------------------
__global__ void __launch_bounds__(256)
scatter_kernel(const float* __restrict__ h,
               const int* __restrict__ esrc,
               const int* __restrict__ edst,
               const float* __restrict__ ew,
               float* __restrict__ out) {
    unsigned gid = blockIdx.x * 256u + threadIdx.x;   // E*64 threads total
    int e = (int)(gid >> 6);
    int c = (int)(gid & 63u) * 4;
    if (e >= NEDGE) return;
    int s = esrc[e], d = edst[e];
    float w = ew[e];
    float4 v = *(const float4*)&h[(size_t)s * DOUT + c];
    float* o = &out[(size_t)d * DOUT + c];
    atomicAdd(o + 0, w * v.x);
    atomicAdd(o + 1, w * v.y);
    atomicAdd(o + 2, w * v.z);
    atomicAdd(o + 3, w * v.w);
}

__global__ void zero_kernel(float* __restrict__ out, int n) {
    int i = blockIdx.x * 256 + threadIdx.x;
    if (i < n) out[i] = 0.0f;
}

__global__ void relu_kernel(float* __restrict__ out, int n) {
    int i = blockIdx.x * 256 + threadIdx.x;
    if (i < n) out[i] = fmaxf(out[i], 0.0f);
}

// ---------------------------------------------------------------------------
extern "C" void kernel_launch(void* const* d_in, const int* in_sizes, int n_in,
                              void* d_out, int out_size, void* d_ws, size_t ws_size,
                              hipStream_t stream) {
    const float* x    = (const float*)d_in[0];
    const float* W    = (const float*)d_in[1];
    const int*   esrc = (const int*)d_in[2];
    const int*   edst = (const int*)d_in[3];
    const float* ew   = (const float*)d_in[4];
    float* out = (float*)d_out;

    float*  h  = (float*)d_ws;                                   // 100000*256*4 = 102.4 MB
    __bf16* Wp = (__bf16*)((char*)d_ws + (size_t)N_NODES * DOUT * sizeof(float)); // +256 KB

    pack_w_kernel<<<(DIN * DOUT + 255) / 256, 256, 0, stream>>>(W, Wp);
    zero_kernel<<<(N_NODES * DOUT + 255) / 256, 256, 0, stream>>>(out, N_NODES * DOUT);
    gemm_dropout_kernel<<<N_NODES / 16, 256, 0, stream>>>(x, Wp, h);
    scatter_kernel<<<(unsigned)((size_t)NEDGE * 64 / 256), 256, 0, stream>>>(h, esrc, edst, ew, out);
    relu_kernel<<<(N_NODES * DOUT + 255) / 256, 256, 0, stream>>>(out, N_NODES * DOUT);
}
